// HGN_12910671692012
// MI455X (gfx1250) — matrix operand order repaired and checked by reading, not compiled, not run on hardware
//
#include <hip/hip_runtime.h>
#include <hip/hip_bf16.h>

typedef __attribute__((ext_vector_type(16))) __bf16 v16bf;
typedef __attribute__((ext_vector_type(8)))  float  v8f;

__device__ __forceinline__ __bf16 f2bf(float f) { return (__bf16)f; }
__device__ __forceinline__ float4 relu4(float4 v) {
    v.x = fmaxf(v.x, 0.f); v.y = fmaxf(v.y, 0.f);
    v.z = fmaxf(v.z, 0.f); v.w = fmaxf(v.w, 0.f);
    return v;
}

template<int ACT>
__device__ __forceinline__ v16bf make_afrag(float4 a0, float4 a1, float4 a2, float4 a3) {
    if (ACT) { a0 = relu4(a0); a1 = relu4(a1); a2 = relu4(a2); a3 = relu4(a3); }
    return (v16bf){
        f2bf(a0.x), f2bf(a0.y), f2bf(a0.z), f2bf(a0.w),
        f2bf(a1.x), f2bf(a1.y), f2bf(a1.z), f2bf(a1.w),
        f2bf(a2.x), f2bf(a2.y), f2bf(a2.z), f2bf(a2.w),
        f2bf(a3.x), f2bf(a3.y), f2bf(a3.z), f2bf(a3.w)};
}

// Guarded (edge-tile) store: one base pointer per tile, advanced by M per row.
__device__ __forceinline__ void store_tiles_guarded(v8f (&acc)[4], const float* bias,
                                                    float* C, int N, int M,
                                                    int row0, int cb, int half, int l16) {
    #pragma unroll
    for (int t = 0; t < 4; ++t) {
        const int ccol = cb * 16 + t * 16 + l16;
        if (ccol < M) {
            const float bv = bias ? bias[ccol] : 0.0f;
            const int r0 = row0 + 8 * half;
            float* Cp = C + (size_t)r0 * M + ccol;
            #pragma unroll
            for (int r = 0; r < 8; ++r) {
                if (r0 + r < N) *Cp = acc[t][r] + bv;
                Cp += M;
            }
        }
    }
}

// ---------------------------------------------------------------------------
// Weight pack: W[K,M] f32 -> bf16 tiles in WMMA B-fragment lane layout,
// zero-padded to Kp=ceil32(K), Mp=ceil64(M). frag idx = (ks*tilesM + c)*32 + l.
// ---------------------------------------------------------------------------
__global__ void __launch_bounds__(256)
hgn_pack_w(const float* __restrict__ W, __bf16* __restrict__ Wpk, int K, int M) {
    const int Kp = (K + 31) & ~31;
    const int Mp = (M + 63) & ~63;
    const int tilesM = Mp >> 4;
    const int total = (Kp >> 5) * tilesM * 32;
    int idx = blockIdx.x * blockDim.x + threadIdx.x;
    if (idx >= total) return;
    const int lane = idx & 31;
    const int tile = idx >> 5;
    const int ks   = tile / tilesM;
    const int c    = tile % tilesM;
    const int half = lane >> 4;
    const int l16  = lane & 15;
    const int col  = c * 16 + l16;
    const int colc = col < M ? col : 0;
    v16bf frag;
    #pragma unroll
    for (int j = 0; j < 16; ++j) {
        int kk  = ks * 32 + 8 * half + (j < 8 ? j : j + 8);
        int kkc = kk < K ? kk : 0;
        float f = W[(size_t)kkc * M + colc];
        frag[j] = f2bf((kk < K && col < M) ? f : 0.0f);
    }
    ((v16bf*)Wpk)[tile * 32 + lane] = frag;
}

// ---------------------------------------------------------------------------
// Fast GEMM (K % 32 == 0): one wave = 16 rows x 64 cols, software-pipelined.
// Full tiles: LDS-transposed epilogue -> 8x global_store_b128 per lane.
// ---------------------------------------------------------------------------
template<int ACT>
__global__ void __launch_bounds__(256)
hgn_gemm_fast(const float* __restrict__ A, const __bf16* __restrict__ Wpk,
              const float* __restrict__ bias, float* __restrict__ C,
              int N, int K, int M) {
    const int lane = threadIdx.x & 31;
    const int wave = threadIdx.x >> 5;
    const int row0 = (blockIdx.x * 8 + wave) * 16;
    if (row0 >= N) return;                    // wave-uniform: EXEC all-1s at WMMA
    const int cb   = blockIdx.y * 4;
    const int half = lane >> 4;
    const int l16  = lane & 15;
    const int tilesM  = ((M + 63) & ~63) >> 4;
    const int kstride = tilesM * 32;          // v16bf frags per K-step
    const int steps   = K >> 5;

    int arow = row0 + l16;
    if (arow >= N) arow = N - 1;              // clamp: loads legal, stores guarded
    const float* Arow = A + (size_t)arow * K + 8 * half;
    const v16bf* Wp   = (const v16bf*)Wpk + cb * 32 + lane;

    v8f acc[4];
    #pragma unroll
    for (int t = 0; t < 4; ++t) acc[t] = (v8f){0.f,0.f,0.f,0.f,0.f,0.f,0.f,0.f};

    // prologue: loads for K-step 0
    float4 a0 = *(const float4*)(Arow + 0);
    float4 a1 = *(const float4*)(Arow + 4);
    float4 a2 = *(const float4*)(Arow + 16);
    float4 a3 = *(const float4*)(Arow + 20);
    v16bf  b0 = Wp[0], b1 = Wp[32], b2 = Wp[64], b3 = Wp[96];

    for (int ks = 0; ks < steps; ++ks) {
        v16bf afrag = make_afrag<ACT>(a0, a1, a2, a3);
        v16bf c0 = b0, c1 = b1, c2 = b2, c3 = b3;
        if (ks + 1 < steps) {                 // uniform: next step's loads in flight
            const float* An = Arow + (ks + 1) * 32;
            a0 = *(const float4*)(An + 0);
            a1 = *(const float4*)(An + 4);
            a2 = *(const float4*)(An + 16);
            a3 = *(const float4*)(An + 20);
            const v16bf* Wn = Wp + (size_t)(ks + 1) * kstride;
            b0 = Wn[0]; b1 = Wn[32]; b2 = Wn[64]; b3 = Wn[96];
        }
        acc[0] = __builtin_amdgcn_wmma_f32_16x16x32_bf16(false, afrag, false, c0, (short)0, acc[0], false, false);
        acc[1] = __builtin_amdgcn_wmma_f32_16x16x32_bf16(false, afrag, false, c1, (short)0, acc[1], false, false);
        acc[2] = __builtin_amdgcn_wmma_f32_16x16x32_bf16(false, afrag, false, c2, (short)0, acc[2], false, false);
        acc[3] = __builtin_amdgcn_wmma_f32_16x16x32_bf16(false, afrag, false, c3, (short)0, acc[3], false, false);
    }

    const int colBase = cb * 16;
    if (row0 + 16 <= N && colBase + 64 <= M) {
        // ---- full-tile epilogue: LDS transpose, vectorized stores ----
        __shared__ __align__(16) float lds[8 * 1088];   // per-wave 16 rows x 68 stride
        float* Ls = lds + wave * 1088;
        float bv[4];
        #pragma unroll
        for (int t = 0; t < 4; ++t) bv[t] = bias ? bias[colBase + t * 16 + l16] : 0.0f;
        #pragma unroll
        for (int t = 0; t < 4; ++t)
            #pragma unroll
            for (int r = 0; r < 8; ++r)
                Ls[(8 * half + r) * 68 + t * 16 + l16] = acc[t][r] + bv[t];
        float* Cbase = C + (size_t)row0 * M + colBase;
        #pragma unroll
        for (int i = 0; i < 8; ++i) {
            const int q    = lane + 32 * i;   // quad index 0..255
            const int row  = q >> 4;
            const int colq = (q & 15) * 4;
            float4 v = *(const float4*)(Ls + row * 68 + colq);
            *(float4*)(Cbase + (size_t)(row * M) + colq) = v;
        }
    } else {
        store_tiles_guarded(acc, bias, C, N, M, row0, cb, half, l16);
    }
}

// ---------------------------------------------------------------------------
// Small-K GEMM (K < 32, first layer only): one zero-padded K-step.
// ---------------------------------------------------------------------------
__global__ void __launch_bounds__(256)
hgn_gemm_small(const float* __restrict__ A, const __bf16* __restrict__ Wpk,
               const float* __restrict__ bias, float* __restrict__ C,
               int N, int K, int M) {
    const int lane = threadIdx.x & 31;
    const int wave = threadIdx.x >> 5;
    const int row0 = (blockIdx.x * 8 + wave) * 16;
    if (row0 >= N) return;
    const int cb   = blockIdx.y * 4;
    const int half = lane >> 4;
    const int l16  = lane & 15;

    int arow = row0 + l16;
    if (arow >= N) arow = N - 1;
    const float* Arow = A + (size_t)arow * K;

    v16bf afrag;
    #pragma unroll
    for (int i = 0; i < 8; ++i) {
        int ka = 8 * half + i;
        int kb = ka + 16;
        float fa = Arow[ka < K ? ka : 0];
        float fb = Arow[kb < K ? kb : 0];
        afrag[i]     = f2bf(ka < K ? fa : 0.0f);
        afrag[i + 8] = f2bf(kb < K ? fb : 0.0f);
    }

    const v16bf* Wp = (const v16bf*)Wpk + cb * 32 + lane;
    v8f acc[4];
    #pragma unroll
    for (int t = 0; t < 4; ++t) {
        v16bf bfrag = Wp[t * 32];
        v8f z = (v8f){0.f,0.f,0.f,0.f,0.f,0.f,0.f,0.f};
        acc[t] = __builtin_amdgcn_wmma_f32_16x16x32_bf16(false, afrag, false, bfrag, (short)0, z, false, false);
    }
    store_tiles_guarded(acc, bias, C, N, M, row0, cb, half, l16);
}

// out[n, :] = bias[:]   (C = 128)
__global__ void __launch_bounds__(256)
hgn_init_bias(float* __restrict__ out, const float* __restrict__ bias, long long total) {
    long long i = (long long)blockIdx.x * blockDim.x + threadIdx.x;
    if (i < total) out[i] = bias[i & 127];
}

// One wave per edge: out[dst[e], :] += tmp[src[e], :] (128-wide latent rows).
__global__ void __launch_bounds__(256)
hgn_scatter_add(const float* __restrict__ tmp, const int* __restrict__ src,
                const int* __restrict__ dst, float* __restrict__ out, int E) {
    const int e = blockIdx.x * 8 + (threadIdx.x >> 5);
    if (e >= E) return;
    const int lane = threadIdx.x & 31;
    const int s = src[e];
    const int d = dst[e];
    float4 v = ((const float4*)(tmp + (size_t)s * 128))[lane];
    float* drow = out + (size_t)d * 128 + lane * 4;
    unsafeAtomicAdd(drow + 0, v.x);
    unsafeAtomicAdd(drow + 1, v.y);
    unsafeAtomicAdd(drow + 2, v.z);
    unsafeAtomicAdd(drow + 3, v.w);
}

extern "C" void kernel_launch(void* const* d_in, const int* in_sizes, int n_in,
                              void* d_out, int out_size, void* d_ws, size_t ws_size,
                              hipStream_t stream) {
    const float* x      = (const float*)d_in[0];
    const int*   edge   = (const int*)  d_in[1];
    const float* enc_W0 = (const float*)d_in[2];
    const float* enc_b0 = (const float*)d_in[3];
    const float* enc_Wh = (const float*)d_in[4];
    const float* enc_bh = (const float*)d_in[5];
    const float* enc_Wl = (const float*)d_in[6];
    const float* enc_bl = (const float*)d_in[7];
    const float* gnc_W  = (const float*)d_in[8];
    const float* gnc_b  = (const float*)d_in[9];
    const float* dec_W0 = (const float*)d_in[10];
    const float* dec_b0 = (const float*)d_in[11];
    const float* dec_Wh = (const float*)d_in[12];
    const float* dec_bh = (const float*)d_in[13];
    const float* dec_Wl = (const float*)d_in[14];
    const float* dec_bl = (const float*)d_in[15];

    const int N = in_sizes[0] / 7;
    const int E = in_sizes[1] / 2;
    const int* esrc = edge;
    const int* edst = edge + E;

    float*  buf0 = (float*)d_ws;
    float*  buf1 = buf0 + (size_t)N * 256;
    __bf16* wpk  = (__bf16*)(buf1 + (size_t)N * 256);   // <=128KB packed weights

    auto gemm = [&](const float* A, const float* W, const float* b, float* Co,
                    int n, int k, int m, int actIn) {
        const int Kp = (k + 31) & ~31, Mp = (m + 63) & ~63;
        const int packThreads = (Kp >> 5) * (Mp >> 4) * 32;
        hgn_pack_w<<<(packThreads + 255) / 256, 256, 0, stream>>>(W, wpk, k, m);
        dim3 grid((n + 127) / 128, (m + 63) / 64);
        if ((k & 31) == 0) {
            if (actIn) hgn_gemm_fast<1><<<grid, 256, 0, stream>>>(A, wpk, b, Co, n, k, m);
            else       hgn_gemm_fast<0><<<grid, 256, 0, stream>>>(A, wpk, b, Co, n, k, m);
        } else {
            hgn_gemm_small<<<grid, 256, 0, stream>>>(A, wpk, b, Co, n, k, m);
        }
    };

    // ---- encoder (ReLU deferred to the input side of the consuming GEMM) ----
    gemm(x,    enc_W0,           enc_b0,       buf0, N,   7, 256, 0);
    gemm(buf0, enc_Wh,           enc_bh,       buf1, N, 256, 256, 1);
    gemm(buf1, enc_Wh + 256*256, enc_bh + 256, buf0, N, 256, 256, 1);
    gemm(buf0, enc_Wl,           enc_bl,       buf1, N, 256, 128, 1);

    // ---- GCN layers ----
    float* h     = buf1;
    float* other = buf0;
    const long long totalH = (long long)N * 128;
    const int initBlocks = (int)((totalH + 255) / 256);
    for (int i = 0; i < 4; ++i) {
        gemm(h, gnc_W + (size_t)i * 128 * 128, nullptr, other, N, 128, 128, i == 0 ? 0 : 1);
        hgn_init_bias<<<initBlocks, 256, 0, stream>>>(h, gnc_b + i * 128, totalH);
        hgn_scatter_add<<<(E + 7) / 8, 256, 0, stream>>>(other, esrc, edst, h, E);
    }

    // ---- decoder ----
    gemm(h,     dec_W0,           dec_b0,       other, N, 128, 256, 1);
    gemm(other, dec_Wh,           dec_bh,       h,     N, 256, 256, 1);
    gemm(h,     dec_Wh + 256*256, dec_bh + 256, other, N, 256, 256, 1);
    gemm(other, dec_Wl,           dec_bl, (float*)d_out, N, 256, 4, 1);
}
